// Seq2SeqAttention_57543971832467
// MI455X (gfx1250) — matrix operand order repaired and checked by reading, not compiled, not run on hardware
//
#include <hip/hip_runtime.h>
#include <hip/hip_bf16.h>
#include <cmath>

// ---------------------------------------------------------------------------
// Seq2Seq + attention for MI455X (gfx1250, wave32).
// All GEMMs: V_WMMA_F32_16X16X32_BF16 (bf16 A/B, f32 accumulate), with
// 32x64 register blocking per wave (8 WMMAs per K-chunk, 2x A / 4x B reuse).
// ---------------------------------------------------------------------------

typedef __bf16 bf16;
typedef __attribute__((ext_vector_type(16))) __bf16 v16bf;
typedef __attribute__((ext_vector_type(8)))  float  v8f;

#define Tn 512
#define Bn 128
#define In 256
#define Hn 1024
#define On 256
#define LEN_OUT 30   // reference len_out (device scalar; fixed by the reference)

static __device__ __forceinline__ float sigm(float x) { return 1.0f / (1.0f + __expf(-x)); }

union frag_u { uint4 u[2]; v16bf v; };

// A fragment: 16x32 bf16, row-major A (lda = K). ISA layout:
// lane L (m = L&15): VGPR j<4 -> K = (L>=16?8:0) + 2j  ; VGPR j>=4 -> +16.
// => per lane two contiguous 8-element (16B) runs at k0+kh and k0+kh+16.
static __device__ __forceinline__ v16bf load_frag_a(const bf16* __restrict__ A, int lda,
                                                    int row0, int k0, int lane) {
  int m  = lane & 15;
  int kh = (lane >> 4) << 3;
  const bf16* p = A + (size_t)(row0 + m) * lda + k0 + kh;
  frag_u t;
  t.u[0] = *reinterpret_cast<const uint4*>(p);
  t.u[1] = *reinterpret_cast<const uint4*>(p + 16);
  return t.v;
}

// B fragment: 32x16 bf16 where B[k][n] = W[n][k], W row-major (ldw = K).
// lane L (n = L&15): same K layout as A: VGPR j<4 -> K = (L>=16?8:0) + 2j ; j>=4 -> +16.
// => per lane two contiguous 8-element (16B) runs at k0+kh and k0+kh+16.
static __device__ __forceinline__ v16bf load_frag_b(const bf16* __restrict__ W, int ldw,
                                                    int col0, int k0, int lane) {
  int n  = lane & 15;
  int kh = (lane >> 4) << 3;
  const bf16* p = W + (size_t)(col0 + n) * ldw + k0 + kh;
  frag_u t;
  t.u[0] = *reinterpret_cast<const uint4*>(p);
  t.u[1] = *reinterpret_cast<const uint4*>(p + 16);
  return t.v;
}

#define WMMA_BF16(a, b, c) \
  __builtin_amdgcn_wmma_f32_16x16x32_bf16(false, (a), false, (b), (short)0, (c), false, false)

// Accumulate one K-segment into the 2x4 accumulator block.
static __device__ __forceinline__ void gemm_seg(const bf16* __restrict__ A,
                                                const bf16* __restrict__ W, int K,
                                                int row0, int col0, int lane,
                                                v8f acc[2][4]) {
  for (int k0 = 0; k0 < K; k0 += 32) {
    v16bf a0 = load_frag_a(A, K, row0,      k0, lane);
    v16bf a1 = load_frag_a(A, K, row0 + 16, k0, lane);
    v16bf b0 = load_frag_b(W, K, col0,      k0, lane);
    v16bf b1 = load_frag_b(W, K, col0 + 16, k0, lane);
    v16bf b2 = load_frag_b(W, K, col0 + 32, k0, lane);
    v16bf b3 = load_frag_b(W, K, col0 + 48, k0, lane);
    acc[0][0] = WMMA_BF16(a0, b0, acc[0][0]);
    acc[0][1] = WMMA_BF16(a0, b1, acc[0][1]);
    acc[0][2] = WMMA_BF16(a0, b2, acc[0][2]);
    acc[0][3] = WMMA_BF16(a0, b3, acc[0][3]);
    acc[1][0] = WMMA_BF16(a1, b0, acc[1][0]);
    acc[1][1] = WMMA_BF16(a1, b1, acc[1][1]);
    acc[1][2] = WMMA_BF16(a1, b2, acc[1][2]);
    acc[1][3] = WMMA_BF16(a1, b3, acc[1][3]);
  }
}

// Generic tiled GEMM: C[M,N] = act( A1[M,K1]*W1[N,K1]^T + A2[M,K2]*W2[N,K2]^T + bias ).
// One wave32 per 32x64 output block (M % 32 == 0, N % 64 == 0 for all call sites).
__global__ void gemm_wmma_bf16(const bf16* __restrict__ A1, const bf16* __restrict__ W1, int K1,
                               const bf16* __restrict__ A2, const bf16* __restrict__ W2, int K2,
                               const float* __restrict__ bias,
                               float* __restrict__ Cf, bf16* __restrict__ Cbf,
                               int M, int N, int act) {
  const int lane    = threadIdx.x & 31;
  const int wave    = (blockIdx.x * blockDim.x + threadIdx.x) >> 5;
  const int tiles_n = N >> 6;                 // 64-wide blocks
  const int total   = (M >> 5) * tiles_n;     // 32-tall blocks
  if (wave >= total) return;
  const int tile_m = wave / tiles_n;
  const int tile_n = wave - tile_m * tiles_n;
  const int row0 = tile_m << 5, col0 = tile_n << 6;

  v8f acc[2][4];
#pragma unroll
  for (int i = 0; i < 2; ++i)
#pragma unroll
    for (int j = 0; j < 4; ++j) acc[i][j] = (v8f){0.f, 0.f, 0.f, 0.f, 0.f, 0.f, 0.f, 0.f};

  gemm_seg(A1, W1, K1, row0, col0, lane, acc);
  if (A2) gemm_seg(A2, W2, K2, row0, col0, lane, acc);

  // C/D layout: lane L: n = L&15, m = e + 8*(L>=16), e = VGPR index 0..7
  const int n  = lane & 15;
  const int mb = (lane >> 4) << 3;
#pragma unroll
  for (int j = 0; j < 4; ++j) {
    const int cn = col0 + (j << 4) + n;
    const float bv = bias ? bias[cn] : 0.0f;
#pragma unroll
    for (int i = 0; i < 2; ++i) {
#pragma unroll
      for (int e = 0; e < 8; ++e) {
        float x = acc[i][j][e] + bv;
        if (act == 1) x = fmaxf(x, 0.0f);
        size_t o = (size_t)(row0 + (i << 4) + mb + e) * N + cn;
        if (Cf)  Cf[o]  = x;
        if (Cbf) Cbf[o] = (bf16)x;
      }
    }
  }
}

// LSTM pointwise: gates[B,4H] (i|f|g|o) -> h,c updates; mirror h to bf16 (+optional 2nd copy).
__global__ void lstm_elem(const float* __restrict__ gates, float* __restrict__ h,
                          float* __restrict__ c, bf16* __restrict__ h_bf,
                          bf16* __restrict__ h_bf2) {
  int idx = blockIdx.x * blockDim.x + threadIdx.x;
  if (idx >= Bn * Hn) return;
  int b = idx >> 10, g = idx & (Hn - 1);
  const float* gr = gates + (size_t)b * 4 * Hn;
  float ig = sigm(gr[g]);
  float fg = sigm(gr[Hn + g]);
  float gg = tanhf(gr[2 * Hn + g]);
  float og = sigm(gr[3 * Hn + g]);
  float cn = fg * c[idx] + ig * gg;
  float hn = og * tanhf(cn);
  c[idx] = cn;
  h[idx] = hn;
  bf16 hb = (bf16)hn;
  h_bf[idx] = hb;
  if (h_bf2) h_bf2[idx] = hb;
}

// scores[t,b] = sum_g v[g] * tanh(hW[b,g] + ctx_proj[t,b,g]); one wave per (t,b).
// ctx_proj is a 268MB one-shot stream per decode step -> non-temporal loads.
__global__ void attn_scores(const float* __restrict__ hW, const float* __restrict__ ctx_proj,
                            const float* __restrict__ v, float* __restrict__ scores) {
  int wave = (blockIdx.x * blockDim.x + threadIdx.x) >> 5;
  int lane = threadIdx.x & 31;
  if (wave >= Tn * Bn) return;
  int t = wave >> 7, b = wave & (Bn - 1);
  const float* cp = ctx_proj + (size_t)(t * Bn + b) * Hn;
  const float* hw = hW + (size_t)b * Hn;
  float acc = 0.0f;
  for (int g = lane; g < Hn; g += 32)
    acc += v[g] * tanhf(hw[g] + __builtin_nontemporal_load(cp + g));
  for (int off = 16; off; off >>= 1) acc += __shfl_xor(acc, off, 32);
  if (lane == 0) scores[t * Bn + b] = acc;
}

// softmax over t (T=512) per batch column b; one wave per b.
__global__ void attn_softmax(const float* __restrict__ scores, float* __restrict__ w) {
  int wave = (blockIdx.x * blockDim.x + threadIdx.x) >> 5;
  int lane = threadIdx.x & 31;
  if (wave >= Bn) return;
  int b = wave;
  float m = -3.402823466e38f;
  for (int t = lane; t < Tn; t += 32) m = fmaxf(m, scores[t * Bn + b]);
  for (int off = 16; off; off >>= 1) m = fmaxf(m, __shfl_xor(m, off, 32));
  float s = 0.0f;
  for (int t = lane; t < Tn; t += 32) s += __expf(scores[t * Bn + b] - m);
  for (int off = 16; off; off >>= 1) s += __shfl_xor(s, off, 32);
  float inv = 1.0f / s;
  for (int t = lane; t < Tn; t += 32)
    w[t * Bn + b] = __expf(scores[t * Bn + b] - m) * inv;
}

// context[b,h] = sum_t w[t,b] * ctx_seq_bf[t,b,h]  (ctx_seq bf16 is L2-resident, 134MB)
__global__ void attn_context(const bf16* __restrict__ ctx_seq, const float* __restrict__ w,
                             bf16* __restrict__ ctx_bf) {
  int idx = blockIdx.x * blockDim.x + threadIdx.x;
  if (idx >= Bn * Hn) return;
  int b = idx >> 10, hi = idx & (Hn - 1);
  float acc = 0.0f;
  for (int t = 0; t < Tn; ++t)
    acc += w[t * Bn + b] * (float)ctx_seq[((size_t)t * Bn + b) * Hn + hi];
  ctx_bf[idx] = (bf16)acc;
}

__global__ void cvt_f32_bf16(const float* __restrict__ s, bf16* __restrict__ d, size_t n) {
  size_t i = (size_t)blockIdx.x * blockDim.x + threadIdx.x;
  if (i < n) d[i] = (bf16)s[i];
}

// Split W_attn[H,2H] into W_h[H,H] (k<H) and W_c[H,H] (k>=H), bf16.
__global__ void split_attn(const float* __restrict__ Wa, bf16* __restrict__ Wh,
                           bf16* __restrict__ Wc) {
  int idx = blockIdx.x * blockDim.x + threadIdx.x;
  if (idx >= Hn * Hn) return;
  int g = idx >> 10, k = idx & (Hn - 1);
  Wh[idx] = (bf16)Wa[(size_t)g * 2 * Hn + k];
  Wc[idx] = (bf16)Wa[(size_t)g * 2 * Hn + Hn + k];
}

__global__ void add_vecs(const float* __restrict__ a, const float* __restrict__ b,
                         float* __restrict__ o, int n) {
  int i = blockIdx.x * blockDim.x + threadIdx.x;
  if (i < n) o[i] = a[i] + b[i];
}

__global__ void zero_f32(float* __restrict__ p, size_t n) {
  size_t i = (size_t)blockIdx.x * blockDim.x + threadIdx.x;
  if (i < n) p[i] = 0.0f;
}

static inline int gemm_blocks(int M, int N) {
  int waves = (M >> 5) * (N >> 6);   // one wave per 32x64 block
  return (waves + 7) / 8;            // 8 waves (256 threads) per block
}

extern "C" void kernel_launch(void* const* d_in, const int* in_sizes, int n_in,
                              void* d_out, int out_size, void* d_ws, size_t ws_size,
                              hipStream_t stream) {
  const float* input_seq = (const float*)d_in[0];
  const float* W_ih_enc  = (const float*)d_in[1];
  const float* W_hh_enc  = (const float*)d_in[2];
  const float* b_ih_enc  = (const float*)d_in[3];
  const float* b_hh_enc  = (const float*)d_in[4];
  const float* W_attn    = (const float*)d_in[5];
  const float* b_attn    = (const float*)d_in[6];
  const float* v_vec     = (const float*)d_in[7];
  const float* W_ih_dec  = (const float*)d_in[8];
  const float* W_hh_dec  = (const float*)d_in[9];
  const float* b_ih_dec  = (const float*)d_in[10];
  const float* b_hh_dec  = (const float*)d_in[11];
  const float* W_dec     = (const float*)d_in[12];
  const float* b_dec     = (const float*)d_in[13];
  const float* W_out     = (const float*)d_in[14];
  const float* b_out     = (const float*)d_in[15];
  (void)in_sizes; (void)n_in; (void)out_size; (void)ws_size;

  // ---- workspace carve-up (256B aligned) ----
  char* ws = (char*)d_ws;
  size_t off = 0;
  auto carve = [&](size_t bytes) -> char* {
    char* p = ws + off;
    off = (off + bytes + 255) & ~(size_t)255;
    return p;
  };
  bf16* x_bf     = (bf16*)carve((size_t)Tn * Bn * In * 2);        // 33.5 MB
  bf16* Wihe_bf  = (bf16*)carve((size_t)4 * Hn * In * 2);
  bf16* Whhe_bf  = (bf16*)carve((size_t)4 * Hn * Hn * 2);
  bf16* Wh_bf    = (bf16*)carve((size_t)Hn * Hn * 2);
  bf16* Wc_bf    = (bf16*)carve((size_t)Hn * Hn * 2);
  bf16* Wihd_bf  = (bf16*)carve((size_t)4 * Hn * Hn * 2);
  bf16* Whhd_bf  = (bf16*)carve((size_t)4 * Hn * Hn * 2);
  bf16* Wdec_bf  = (bf16*)carve((size_t)Hn * Hn * 2);
  bf16* Wout_bf  = (bf16*)carve((size_t)On * Hn * 2);
  float* b_enc   = (float*)carve((size_t)4 * Hn * 4);
  float* b_decC  = (float*)carve((size_t)4 * Hn * 4);
  bf16* ctxseq   = (bf16*)carve((size_t)Tn * Bn * Hn * 2);        // 134 MB (L2-resident)
  float* ctxproj = (float*)carve((size_t)Tn * Bn * Hn * 4);       // 268 MB (NT streamed)
  float* gates   = (float*)carve((size_t)Bn * 4 * Hn * 4);
  float* h_f     = (float*)carve((size_t)Bn * Hn * 4);
  float* c_f     = (float*)carve((size_t)Bn * Hn * 4);
  bf16* h_bf     = (bf16*)carve((size_t)Bn * Hn * 2);
  float* hW      = (float*)carve((size_t)Bn * Hn * 4);
  float* scores  = (float*)carve((size_t)Tn * Bn * 4);
  float* attnw   = (float*)carve((size_t)Tn * Bn * 4);
  bf16* ctx_bf   = (bf16*)carve((size_t)Bn * Hn * 2);
  bf16* out1_bf  = (bf16*)carve((size_t)Bn * Hn * 2);

  // ---- one-time (per launch) packs: f32 -> bf16, combined biases, zero state ----
  auto cvt = [&](const float* s, bf16* d, size_t n) {
    cvt_f32_bf16<<<(unsigned)((n + 255) / 256), 256, 0, stream>>>(s, d, n);
  };
  cvt(input_seq, x_bf, (size_t)Tn * Bn * In);
  cvt(W_ih_enc, Wihe_bf, (size_t)4 * Hn * In);
  cvt(W_hh_enc, Whhe_bf, (size_t)4 * Hn * Hn);
  cvt(W_ih_dec, Wihd_bf, (size_t)4 * Hn * Hn);
  cvt(W_hh_dec, Whhd_bf, (size_t)4 * Hn * Hn);
  cvt(W_dec,    Wdec_bf, (size_t)Hn * Hn);
  cvt(W_out,    Wout_bf, (size_t)On * Hn);
  split_attn<<<(Hn * Hn) / 256, 256, 0, stream>>>(W_attn, Wh_bf, Wc_bf);
  add_vecs<<<16, 256, 0, stream>>>(b_ih_enc, b_hh_enc, b_enc, 4 * Hn);
  add_vecs<<<16, 256, 0, stream>>>(b_ih_dec, b_hh_dec, b_decC, 4 * Hn);
  zero_f32<<<(Bn * Hn) / 256, 256, 0, stream>>>(h_f, (size_t)Bn * Hn);
  zero_f32<<<(Bn * Hn) / 256, 256, 0, stream>>>(c_f, (size_t)Bn * Hn);
  zero_f32<<<(Bn * Hn / 2) / 256, 256, 0, stream>>>((float*)h_bf, (size_t)Bn * Hn / 2);

  // ---- encoder: 512 sequential steps; gates = [x_t|h] @ [W_ih;W_hh]^T + b ----
  for (int t = 0; t < Tn; ++t) {
    gemm_wmma_bf16<<<gemm_blocks(Bn, 4 * Hn), 256, 0, stream>>>(
        x_bf + (size_t)t * Bn * In, Wihe_bf, In,
        h_bf, Whhe_bf, Hn,
        b_enc, gates, (bf16*)nullptr, Bn, 4 * Hn, 0);
    lstm_elem<<<(Bn * Hn) / 256, 256, 0, stream>>>(
        gates, h_f, c_f, h_bf, ctxseq + (size_t)t * Bn * Hn);
  }

  // ---- ctx_proj[t,b,g] = ctx_seq @ W_c^T + b_attn (one 65536x1024x1024 GEMM) ----
  gemm_wmma_bf16<<<gemm_blocks(Tn * Bn, Hn), 256, 0, stream>>>(
      ctxseq, Wc_bf, Hn, (bf16*)nullptr, (bf16*)nullptr, 0,
      b_attn, ctxproj, (bf16*)nullptr, Tn * Bn, Hn, 0);

  // ---- decoder: h starts at h_enc (already in h_f/h_bf), c restarts at 0 ----
  zero_f32<<<(Bn * Hn) / 256, 256, 0, stream>>>(c_f, (size_t)Bn * Hn);
  for (int s = 0; s < LEN_OUT; ++s) {
    gemm_wmma_bf16<<<gemm_blocks(Bn, Hn), 256, 0, stream>>>(          // hW = h @ W_h^T
        h_bf, Wh_bf, Hn, (bf16*)nullptr, (bf16*)nullptr, 0,
        (const float*)nullptr, hW, (bf16*)nullptr, Bn, Hn, 0);
    attn_scores<<<(Tn * Bn * 32) / 256, 256, 0, stream>>>(hW, ctxproj, v_vec, scores);
    attn_softmax<<<(Bn * 32) / 256, 256, 0, stream>>>(scores, attnw);
    attn_context<<<(Bn * Hn) / 256, 256, 0, stream>>>(ctxseq, attnw, ctx_bf);
    gemm_wmma_bf16<<<gemm_blocks(Bn, 4 * Hn), 256, 0, stream>>>(      // decoder LSTM gates
        ctx_bf, Wihd_bf, Hn, h_bf, Whhd_bf, Hn,
        b_decC, gates, (bf16*)nullptr, Bn, 4 * Hn, 0);
    lstm_elem<<<(Bn * Hn) / 256, 256, 0, stream>>>(gates, h_f, c_f, h_bf, (bf16*)nullptr);
    gemm_wmma_bf16<<<gemm_blocks(Bn, Hn), 256, 0, stream>>>(          // relu(h @ W_dec^T + b)
        h_bf, Wdec_bf, Hn, (bf16*)nullptr, (bf16*)nullptr, 0,
        b_dec, (float*)nullptr, out1_bf, Bn, Hn, 1);
    gemm_wmma_bf16<<<gemm_blocks(Bn, On), 256, 0, stream>>>(          // out @ W_out^T + b
        out1_bf, Wout_bf, Hn, (bf16*)nullptr, (bf16*)nullptr, 0,
        b_out, (float*)d_out + (size_t)s * Bn * On, (bf16*)nullptr, Bn, On, 0);
  }
}
